// AttentionBlock_45938970198761
// MI455X (gfx1250) — matrix operand order repaired and checked by reading, not compiled, not run on hardware
//
#include <hip/hip_runtime.h>

typedef _Float16 half_t;
typedef __attribute__((ext_vector_type(16))) _Float16 v16h;
typedef __attribute__((ext_vector_type(8)))  _Float16 v8h;
typedef __attribute__((ext_vector_type(8)))  float    v8f;

#define CDIM   256
#define HWDIM  4096
#define NBATCH 4
#define NGROUP 8
#define CPG    32
#define PLD    48          // LDS row stride (halves) for P tile: 96B, 16B aligned
#define NPITCH 68          // LDS row pitch (floats) for gn tile: 272B, 16B aligned
#define ATT_SCALE 0.0625f  // 1/sqrt(256)

// Async global->LDS copy path (gfx1250); guarded so absence degrades gracefully.
#if defined(__gfx1250__) && __has_builtin(__builtin_amdgcn_global_load_async_to_lds_b128) && __has_builtin(__builtin_amdgcn_s_wait_asynccnt)
#define HAVE_ASYNC_LDS 1
// Exact param types per hipcc diagnostic: (int4 AS1*, int4 AS3*, imm int, imm int)
typedef int v4i_vs __attribute__((vector_size(16)));
typedef __attribute__((address_space(1))) v4i_vs* gas1_i4p;
typedef __attribute__((address_space(3))) v4i_vs* las3_i4p;
#else
#define HAVE_ASYNC_LDS 0
#endif

// ---------------------------------------------------------------------------
// WMMA helpers (gfx1250 CDNA5): D = A(16x32 f16) * B(32x16 f16) + C(16x16 f32)
// ---------------------------------------------------------------------------
static __device__ __forceinline__ v8f wmma16(v16h a, v16h b, v8f c) {
  return __builtin_amdgcn_wmma_f32_16x16x32_f16(false, a, false, b, (short)0, c,
                                                false, false);
}

// A-matrix 16x32 f16 fragment from a row-major matrix (row = M, contiguous K).
// lane L (m=L&15, hs=L>>4): elems 0..7 -> K=hs*8+e, elems 8..15 -> K=16+hs*8+e.
static __device__ __forceinline__ v16h load_a_frag(const half_t* p, int ld,
                                                   int row0, int col0, int lane) {
  const int m = lane & 15, hs = lane >> 4;
  const half_t* r = p + (size_t)(row0 + m) * ld + col0 + hs * 8;
  v8h lo = *(const v8h*)(r);
  v8h hi = *(const v8h*)(r + 16);
  v16h a;
#pragma unroll
  for (int e = 0; e < 8; ++e) { a[e] = lo[e]; a[e + 8] = hi[e]; }
  return a;
}

// B-matrix 32x16 f16 fragment from an N-major ("Bt") matrix: row = N column of B,
// contiguous K. lane L (n=L&15, hs=L>>4): elems 0..7 -> K=hs*8+e, elems 8..15 -> K=16+hs*8+e.
static __device__ __forceinline__ v16h load_b_frag(const half_t* p, int ld,
                                                   int nrow0, int col0, int lane) {
  const int nn = lane & 15, ko = (lane >> 4) * 8;
  const half_t* r = p + (size_t)(nrow0 + nn) * ld + col0 + ko;
  v8h lo = *(const v8h*)(r);
  v8h hi = *(const v8h*)(r + 16);
  v16h b;
#pragma unroll
  for (int e = 0; e < 8; ++e) { b[e] = lo[e]; b[e + 8] = hi[e]; }
  return b;
}

// ---------------------------------------------------------------------------
// GroupNorm: per-(batch,group) mean / rstd
// ---------------------------------------------------------------------------
__global__ __launch_bounds__(256) void gn_stats_kernel(const float* __restrict__ x,
                                                       float* __restrict__ stats) {
  const int bg = blockIdx.x;  // n*8 + g  (channels of a group are contiguous)
  const float* base = x + (size_t)bg * (CPG * HWDIM);
  float s = 0.f, s2 = 0.f;
  for (int idx = threadIdx.x; idx < CPG * HWDIM; idx += 256) {
    float vv = base[idx];
    s += vv; s2 += vv * vv;
  }
  __shared__ float rs[256], rs2[256];
  rs[threadIdx.x] = s; rs2[threadIdx.x] = s2;
  __syncthreads();
  for (int off = 128; off > 0; off >>= 1) {
    if ((int)threadIdx.x < off) {
      rs[threadIdx.x]  += rs[threadIdx.x + off];
      rs2[threadIdx.x] += rs2[threadIdx.x + off];
    }
    __syncthreads();
  }
  if (threadIdx.x == 0) {
    const float inv = 1.0f / (float)(CPG * HWDIM);
    float mean = rs[0] * inv;
    float var  = rs2[0] * inv - mean * mean;
    stats[bg]      = mean;
    stats[32 + bg] = rsqrtf(var + 1e-5f);
  }
}

// Normalize + affine + f16 + transpose to xnT[n][i][c], tiled 64c x 64i via LDS.
// Load phase: coalesced 256B row reads (async global->LDS when available).
// Store phase: contiguous 64B-per-wave f16 writes.
__global__ __launch_bounds__(256) void gn_norm_kernel(const float* __restrict__ x,
                                                      const float* __restrict__ stats,
                                                      const float* __restrict__ gw,
                                                      const float* __restrict__ gb,
                                                      half_t* __restrict__ xnT) {
  __shared__ float tile[64 * NPITCH];
  const int n  = blockIdx.z;
  const int c0 = blockIdx.y * 64;
  const int i0 = blockIdx.x * 64;
  const int tid = threadIdx.x;

  const int il = (tid & 15) * 4;   // 16 float4 segments per 64-wide row
  const int cl = tid >> 4;         // 16 rows per pass, 4 passes
  const float* src = x + ((size_t)n * CDIM + c0) * HWDIM + i0;

#if HAVE_ASYNC_LDS
  for (int cc = cl; cc < 64; cc += 16) {
    __builtin_amdgcn_global_load_async_to_lds_b128(
        (gas1_i4p)(void*)(src + (size_t)cc * HWDIM + il),
        (las3_i4p)(void*)&tile[cc * NPITCH + il],
        0, 0);
  }
  __builtin_amdgcn_s_wait_asynccnt(0);
  __syncthreads();
#else
  for (int cc = cl; cc < 64; cc += 16) {
    *(float4*)&tile[cc * NPITCH + il] =
        *(const float4*)(src + (size_t)cc * HWDIM + il);
  }
  __syncthreads();
#endif

  const int cw = tid & 63;   // c within tile (contiguous in store)
  const int iw = tid >> 6;   // 4 i-rows per pass, 16 passes
  for (int ii = iw; ii < 64; ii += 4) {
    int c = c0 + cw;
    int bg = n * NGROUP + (c >> 5);
    float vv = (tile[cw * NPITCH + ii] - stats[bg]) * stats[32 + bg] * gw[c] + gb[c];
    xnT[((size_t)n * HWDIM + i0 + ii) * CDIM + c] = (half_t)vv;
  }
}

// Convert the four 256x256 weight matrices to f16 (row-major kept)
__global__ __launch_bounds__(256) void wcvt_kernel(const float* __restrict__ wq,
                                                   const float* __restrict__ wk,
                                                   const float* __restrict__ wv,
                                                   const float* __restrict__ wp,
                                                   half_t* __restrict__ wh) {
  int idx = blockIdx.x * 256 + threadIdx.x;   // 4*65536 total
  int m = idx >> 16;
  int r = idx & 65535;
  const float* src = (m == 0) ? wq : (m == 1) ? wk : (m == 2) ? wv : wp;
  wh[idx] = (half_t)src[r];
}

// ---------------------------------------------------------------------------
// QKV GEMMs: [Wq|Wk|Wv](256x256) * Xn(256x4096) per batch, WMMA f16.
// Q stored transposed (i,c), K stored transposed (j,c), V stored natural (c,j).
// ---------------------------------------------------------------------------
__global__ __launch_bounds__(256) void qkv_gemm_kernel(
    const half_t* __restrict__ wh, const float* __restrict__ bq,
    const float* __restrict__ bk, const float* __restrict__ bv,
    const half_t* __restrict__ xnT, half_t* __restrict__ Qt,
    half_t* __restrict__ Kt, half_t* __restrict__ Vn) {
  const int lane = threadIdx.x & 31;
  const int wave = threadIdx.x >> 5;
  const int ib = blockIdx.x;
  const int cb = blockIdx.y;
  const int z  = blockIdx.z;
  const int batch = z / 3;
  const int mat   = z % 3;
  const int c0 = (cb * 8 + wave) * 16;

  const half_t* W = wh + (size_t)mat * CDIM * CDIM;
  const float* bias = (mat == 0) ? bq : (mat == 1) ? bk : bv;

  v16h af[8];
#pragma unroll
  for (int kc = 0; kc < 8; ++kc) af[kc] = load_a_frag(W, CDIM, c0, kc * 32, lane);

  const half_t* xb = xnT + (size_t)batch * HWDIM * CDIM;
  const int nn = lane & 15;
  const int hs = lane >> 4;

  for (int t = 0; t < 4; ++t) {
    const int i0 = ib * 64 + t * 16;
    v8f acc = {};
#pragma unroll
    for (int kc = 0; kc < 8; ++kc)
      acc = wmma16(af[kc], load_b_frag(xb, CDIM, i0, kc * 32, lane), acc);

    if (mat < 2) {
      // transposed store: D[m][n] -> Xt[(i0+n)*C + c0+m]; 16B contiguous per lane
      half_t* dst = (mat == 0 ? Qt : Kt) + (size_t)batch * HWDIM * CDIM +
                    (size_t)(i0 + nn) * CDIM + c0 + hs * 8;
      v8h outv;
#pragma unroll
      for (int e = 0; e < 8; ++e)
        outv[e] = (half_t)(acc[e] + bias[c0 + hs * 8 + e]);
      *(v8h*)dst = outv;
    } else {
      // natural store V[c][j]: per row, 16 lanes cover 32B contiguous
      half_t* vb = Vn + (size_t)batch * CDIM * HWDIM;
#pragma unroll
      for (int e = 0; e < 8; ++e) {
        int m = hs * 8 + e;
        vb[(size_t)(c0 + m) * HWDIM + i0 + nn] = (half_t)(acc[e] + bias[c0 + m]);
      }
    }
  }
}

// ---------------------------------------------------------------------------
// Fused flash attention. One wave owns 16 queries x full d=256.
// S = Qt(i,c) x Kt(j,c)^T via WMMA; online softmax; O += P * V^T via WMMA.
// attn matrix never touches HBM. K/V stream prefetched one j-chunk ahead.
// ---------------------------------------------------------------------------
__global__ __launch_bounds__(128) void attn_flash_kernel(
    const half_t* __restrict__ Qt, const half_t* __restrict__ Kt,
    const half_t* __restrict__ Vn, half_t* __restrict__ Ot) {
  const int lane  = threadIdx.x & 31;
  const int wave  = threadIdx.x >> 5;   // 0..3
  const int batch = blockIdx.y;
  const int i0 = (blockIdx.x * 4 + wave) * 16;

  const half_t* q = Qt + (size_t)batch * HWDIM * CDIM;
  const half_t* k = Kt + (size_t)batch * HWDIM * CDIM;
  const half_t* v = Vn + (size_t)batch * CDIM * HWDIM;

  __shared__ half_t plds[4][16 * PLD];   // per-wave P staging (D-layout -> A-layout)
  half_t* P = plds[wave];

  // Q fragments resident: 8 chunks of K=32 covering d=256
  v16h qa[8];
#pragma unroll
  for (int kc = 0; kc < 8; ++kc) qa[kc] = load_a_frag(q, CDIM, i0, kc * 32, lane);

  v8f zero = {};
  v8f o[16];
#pragma unroll
  for (int ct = 0; ct < 16; ++ct) o[ct] = zero;

  float mrow[8], lrow[8];
#pragma unroll
  for (int e = 0; e < 8; ++e) { mrow[e] = -1e30f; lrow[e] = 0.f; }

  for (int j0 = 0; j0 < HWDIM; j0 += 32) {
    // ---- prefetch next j-chunk (K rows: 512B each = 4 lines; V: 64B col segs)
    {
      const half_t* nk = k + (size_t)(j0 + 32 + lane) * CDIM;
      __builtin_prefetch(nk, 0, 0);
      __builtin_prefetch(nk + 64, 0, 0);
      __builtin_prefetch(nk + 128, 0, 0);
      __builtin_prefetch(nk + 192, 0, 0);
      const half_t* nv = v + (size_t)(lane * 8) * HWDIM + j0 + 32;
#pragma unroll
      for (int r = 0; r < 8; ++r) __builtin_prefetch(nv + (size_t)r * HWDIM, 0, 0);
    }

    // ---- S = Q^T K for two 16-wide j sub-tiles
    v8f s0 = zero, s1 = zero;
#pragma unroll
    for (int kc = 0; kc < 8; ++kc) {
      s0 = wmma16(qa[kc], load_b_frag(k, CDIM, j0,      kc * 32, lane), s0);
      s1 = wmma16(qa[kc], load_b_frag(k, CDIM, j0 + 16, kc * 32, lane), s1);
    }

    // ---- online softmax (row stats via 16-lane half shuffles; elem e = row mbase+e)
    float alpha[8];
    const int nn = lane & 15;
    const int mbase = (lane >> 4) * 8;
#pragma unroll
    for (int e = 0; e < 8; ++e) {
      float a0 = s0[e] * ATT_SCALE;
      float a1 = s1[e] * ATT_SCALE;
      float rm = fmaxf(a0, a1);
      rm = fmaxf(rm, __shfl_xor(rm, 1));
      rm = fmaxf(rm, __shfl_xor(rm, 2));
      rm = fmaxf(rm, __shfl_xor(rm, 4));
      rm = fmaxf(rm, __shfl_xor(rm, 8));
      float nm = fmaxf(mrow[e], rm);
      float al = __expf(mrow[e] - nm);
      float p0 = __expf(a0 - nm);
      float p1 = __expf(a1 - nm);
      float rsum = p0 + p1;
      rsum += __shfl_xor(rsum, 1);
      rsum += __shfl_xor(rsum, 2);
      rsum += __shfl_xor(rsum, 4);
      rsum += __shfl_xor(rsum, 8);
      lrow[e] = lrow[e] * al + rsum;
      mrow[e] = nm;
      alpha[e] = al;
      P[(mbase + e) * PLD + nn]      = (half_t)p0;
      P[(mbase + e) * PLD + 16 + nn] = (half_t)p1;
    }
    __syncthreads();
    v16h pa = load_a_frag(P, PLD, 0, 0, lane);   // P as 16x32 A fragment
    __syncthreads();

    // ---- O = alpha*O + P * V^T  (16 c-tiles cover d=256)
#pragma unroll
    for (int ct = 0; ct < 16; ++ct) {
#pragma unroll
      for (int e = 0; e < 8; ++e) o[ct][e] *= alpha[e];
      o[ct] = wmma16(pa, load_b_frag(v, HWDIM, ct * 16, j0, lane), o[ct]);
    }
  }

  // ---- epilogue: O / l, store transposed Ot[i][c]
  const int nn = lane & 15;
  const int mbase = (lane >> 4) * 8;
  half_t* od = Ot + (size_t)batch * HWDIM * CDIM;
#pragma unroll
  for (int ct = 0; ct < 16; ++ct) {
#pragma unroll
    for (int e = 0; e < 8; ++e) {
      od[(size_t)(i0 + mbase + e) * CDIM + ct * 16 + nn] =
          (half_t)(o[ct][e] / lrow[e]);
    }
  }
}

// ---------------------------------------------------------------------------
// Projection GEMM + bias + residual: out = Wp * O + bp + x   (fp32 out)
// ---------------------------------------------------------------------------
__global__ __launch_bounds__(256) void proj_gemm_kernel(
    const half_t* __restrict__ wph, const float* __restrict__ bp,
    const half_t* __restrict__ Ot, const float* __restrict__ x,
    float* __restrict__ out) {
  const int lane = threadIdx.x & 31;
  const int wave = threadIdx.x >> 5;
  const int ib = blockIdx.x;
  const int cb = blockIdx.y;
  const int batch = blockIdx.z;
  const int c0 = (cb * 8 + wave) * 16;

  v16h af[8];
#pragma unroll
  for (int kc = 0; kc < 8; ++kc) af[kc] = load_a_frag(wph, CDIM, c0, kc * 32, lane);

  const half_t* ob = Ot + (size_t)batch * HWDIM * CDIM;
  const int nn = lane & 15;
  const int hs = lane >> 4;

  for (int t = 0; t < 4; ++t) {
    const int i0 = ib * 64 + t * 16;
    v8f acc = {};
#pragma unroll
    for (int kc = 0; kc < 8; ++kc)
      acc = wmma16(af[kc], load_b_frag(ob, CDIM, i0, kc * 32, lane), acc);
#pragma unroll
    for (int e = 0; e < 8; ++e) {
      int m = hs * 8 + e;
      size_t idx = ((size_t)batch * CDIM + c0 + m) * HWDIM + i0 + nn;
      out[idx] = acc[e] + bp[c0 + m] + x[idx];
    }
  }
}

// ---------------------------------------------------------------------------
extern "C" void kernel_launch(void* const* d_in, const int* in_sizes, int n_in,
                              void* d_out, int out_size, void* d_ws, size_t ws_size,
                              hipStream_t stream) {
  (void)in_sizes; (void)n_in; (void)out_size; (void)ws_size;
  const float* x    = (const float*)d_in[0];
  const float* gn_w = (const float*)d_in[1];
  const float* gn_b = (const float*)d_in[2];
  const float* wq   = (const float*)d_in[3];
  const float* bq   = (const float*)d_in[4];
  const float* wk   = (const float*)d_in[5];
  const float* bk   = (const float*)d_in[6];
  const float* wv   = (const float*)d_in[7];
  const float* bv   = (const float*)d_in[8];
  const float* wp   = (const float*)d_in[9];
  const float* bp   = (const float*)d_in[10];
  float* out = (float*)d_out;

  // workspace layout (~32.5 MB): stats | wh(f16 x4) | xnT/Ot | Qt | Kt | Vn
  uint8_t* ws = (uint8_t*)d_ws;
  float*  stats = (float*)ws;                                        // 64 floats
  half_t* wh    = (half_t*)(ws + 256);                               // 512 KB
  half_t* xnT   = (half_t*)(ws + 256 + 4 * CDIM * CDIM * sizeof(half_t));
  half_t* Qt    = xnT + (size_t)NBATCH * HWDIM * CDIM;
  half_t* Kt    = Qt  + (size_t)NBATCH * HWDIM * CDIM;
  half_t* Vn    = Kt  + (size_t)NBATCH * HWDIM * CDIM;
  half_t* Ot    = xnT;  // xnT dead after QKV GEMM -> reuse for attention output

  gn_stats_kernel<<<NBATCH * NGROUP, 256, 0, stream>>>(x, stats);
  gn_norm_kernel<<<dim3(HWDIM / 64, CDIM / 64, NBATCH), 256, 0, stream>>>(
      x, stats, gn_w, gn_b, xnT);
  wcvt_kernel<<<(4 * CDIM * CDIM) / 256, 256, 0, stream>>>(wq, wk, wv, wp, wh);
  qkv_gemm_kernel<<<dim3(64, 2, 12), 256, 0, stream>>>(wh, bq, bk, bv, xnT, Qt, Kt,
                                                       Vn);
  attn_flash_kernel<<<dim3(64, NBATCH), 128, 0, stream>>>(Qt, Kt, Vn, Ot);
  proj_gemm_kernel<<<dim3(64, 2, NBATCH), 256, 0, stream>>>(wh + 3 * CDIM * CDIM, bp,
                                                            Ot, x, out);
}
